// ConformerEncoder_37769942401603
// MI455X (gfx1250) — matrix operand order repaired and checked by reading, not compiled, not run on hardware
//
#include <hip/hip_runtime.h>
#include <hip/hip_bf16.h>
#include <math.h>

// ---- problem constants ------------------------------------------------------
#define D_    512
#define H_    8
#define DFF_  2048
#define KCONV 31
#define L_    2
#define B_    4
#define T_    1024
#define DK_   64
#define P_    2047
#define PPAD  2048
#define EPS_  1e-5f
#define NROW  (B_*T_)        // 4096 token rows
#define SCALE_ 0.125f        // 1/sqrt(64)

typedef __attribute__((ext_vector_type(16))) _Float16 v16h;
typedef __attribute__((ext_vector_type(8)))  float    v8f;
typedef __attribute__((ext_vector_type(4)))  _Float16 h4;

static __device__ __forceinline__ float sigmoidf_(float x){ return 1.0f/(1.0f+__expf(-x)); }
static __device__ __forceinline__ float siluf_(float x){ return x*sigmoidf_(x); }

// read an aligned f16-pair from LDS into fragment slots {2v, 2v+1}
static __device__ __forceinline__ void frag_set_pair(v16h& f, int v, const _Float16* p) {
  union { unsigned u; _Float16 h[2]; } t;
  t.u = *(const unsigned*)p;
  f[2*v] = t.h[0]; f[2*v+1] = t.h[1];
}

// ============================================================================
// WMMA GEMM: out = res + alpha*act(A[MxK] @ W[KxN] + bias)
//   Requirements: M % 64 == 0, N % 128 == 0, K % 32 == 0.
//   A reads clamped to row <= Arows-1 (branch-free; used by the 2047-row
//   position projection, which runs padded to M=2048).
//   Tile 64x128x32, 8 wave32 waves in 2(M) x 4(N); each wave: 2 A-frags x
//   2 B-frags -> 4 v_wmma_f32_16x16x32_f16 per K-step.
//   LDS double-buffered: next slab's global b128 loads are issued BEFORE the
//   current slab's ds_loads + WMMAs, converted/stored after -> loads overlap
//   matrix math, one barrier per K-step.
// ============================================================================
#define BM 64
#define BN 128
#define BK 32
#define SA 36   // padded LDS row stride in halves (72B: keeps half4 stores 8B-aligned)

__global__ __launch_bounds__(256)
void wmma_gemm_kernel(const float* __restrict__ A, const float* __restrict__ W,
                      const float* __restrict__ bias, const float* __restrict__ res,
                      float* __restrict__ out, int M, int N, int Kd, int Arows,
                      int act, float alpha) {
  __shared__ __align__(16) _Float16 lds_a[2][BM*SA];
  __shared__ __align__(16) _Float16 lds_w[2][BN*SA];   // transposed: [n][k]
  const int tid   = threadIdx.x;
  const int lane  = tid & 31;
  const int wave  = tid >> 5;
  const int wm    = (wave & 1) * 32;   // wave row offset (2 waves along M)
  const int wn    = (wave >> 1) * 32;  // wave col offset (4 waves along N)
  const int n0    = blockIdx.x * BN;
  const int m0    = blockIdx.y * BM;
  const int khalf = lane >> 4;
  const int l16   = lane & 15;

  // per-thread staging coordinates (fixed across K-steps)
  int ar_[2], ac_[2];      // A: 512 float4 chunks, 2 per thread
  #pragma unroll
  for (int i = 0; i < 2; ++i) {
    int e = tid + i*256;
    int r = e >> 3, c4 = (e & 7) << 2;
    int row = m0 + r; row = row < Arows ? row : (Arows - 1);
    ar_[i] = row;  ac_[i] = r*SA + c4;
  }
  int wr_[4], wc_[4];      // W: 1024 float4 chunks, 4 per thread
  #pragma unroll
  for (int i = 0; i < 4; ++i) {
    int e = tid + i*256;
    wr_[i] = e >> 5;               // k row 0..31
    wc_[i] = (e & 31) << 2;        // n col group
  }

  // issue the 6 staging b128 loads for K-slab kt (no conversion yet)
  auto stage_load = [&](int kt, float4* areg, float4* wreg) {
    const int kb0 = kt * BK;
    #pragma unroll
    for (int i = 0; i < 2; ++i)
      areg[i] = *(const float4*)&A[(long)ar_[i]*Kd + kb0 + (ac_[i] % SA)];
    #pragma unroll
    for (int i = 0; i < 4; ++i)
      wreg[i] = *(const float4*)&W[(long)(kb0 + wr_[i])*N + n0 + wc_[i]];
  };
  // convert + store staged registers into LDS buffer `buf`
  auto stage_store = [&](int buf, const float4* areg, const float4* wreg) {
    #pragma unroll
    for (int i = 0; i < 2; ++i) {
      h4 hv;
      hv[0]=(_Float16)areg[i].x; hv[1]=(_Float16)areg[i].y;
      hv[2]=(_Float16)areg[i].z; hv[3]=(_Float16)areg[i].w;
      *(h4*)&lds_a[buf][ac_[i]] = hv;
    }
    #pragma unroll
    for (int i = 0; i < 4; ++i) {
      lds_w[buf][(wc_[i]+0)*SA + wr_[i]] = (_Float16)wreg[i].x;
      lds_w[buf][(wc_[i]+1)*SA + wr_[i]] = (_Float16)wreg[i].y;
      lds_w[buf][(wc_[i]+2)*SA + wr_[i]] = (_Float16)wreg[i].z;
      lds_w[buf][(wc_[i]+3)*SA + wr_[i]] = (_Float16)wreg[i].w;
    }
  };

  v8f c00 = {}, c01 = {}, c10 = {}, c11 = {};
  const int ksteps = Kd / BK;
  float4 areg[2], wreg[4];
  stage_load(0, areg, wreg);
  stage_store(0, areg, wreg);

  for (int kt = 0; kt < ksteps; ++kt) {
    __syncthreads();                       // stores to lds[buf] visible
    const int buf = kt & 1;
    const bool more = (kt + 1) < ksteps;
    if (more) stage_load(kt + 1, areg, wreg);   // loads fly over ds+wmma below

    // A fragments (16x32 f16), rows wm and wm+16
    v16h af0, af1;
    #pragma unroll
    for (int v = 0; v < 8; ++v) {
      int kb = 2*v + (v >= 4 ? 8 : 0) + khalf*8;
      frag_set_pair(af0, v, &lds_a[buf][(wm      + l16)*SA + kb]);
      frag_set_pair(af1, v, &lds_a[buf][(wm + 16 + l16)*SA + kb]);
    }
    // B fragments (32x16 f16), cols wn and wn+16
    v16h bf0, bf1;
    #pragma unroll
    for (int v = 0; v < 8; ++v) {
      int kb = 2*v + (v >= 4 ? 8 : 0) + khalf*8;
      frag_set_pair(bf0, v, &lds_w[buf][(wn      + l16)*SA + kb]);
      frag_set_pair(bf1, v, &lds_w[buf][(wn + 16 + l16)*SA + kb]);
    }
    c00 = __builtin_amdgcn_wmma_f32_16x16x32_f16(false, af0, false, bf0, (short)0, c00, false, false);
    c01 = __builtin_amdgcn_wmma_f32_16x16x32_f16(false, af0, false, bf1, (short)0, c01, false, false);
    c10 = __builtin_amdgcn_wmma_f32_16x16x32_f16(false, af1, false, bf0, (short)0, c10, false, false);
    c11 = __builtin_amdgcn_wmma_f32_16x16x32_f16(false, af1, false, bf1, (short)0, c11, false, false);

    if (more) stage_store(buf ^ 1, areg, wreg);  // other buffer: no barrier needed
  }

  // epilogue: C layout (m = r + 8*khalf, n = l16); no M guard (M % 64 == 0)
  #pragma unroll
  for (int sub = 0; sub < 2; ++sub) {
    #pragma unroll
    for (int cf = 0; cf < 2; ++cf) {
      v8f& cc = sub ? (cf ? c11 : c10) : (cf ? c01 : c00);
      #pragma unroll
      for (int r = 0; r < 8; ++r) {
        int row = m0 + wm + sub*16 + r + 8*khalf;
        int col = n0 + wn + cf*16 + l16;
        float val = cc[r];
        if (bias) val += bias[col];
        if (act == 1) val = siluf_(val);
        long idx = (long)row*N + col;
        out[idx] = res ? (res[idx] + alpha*val) : val;
      }
    }
  }
}

// ============================================================================
// Fused rel-pos flash attention. One wave32 per 16-row tile per (b, head).
// scores(t,s) = SCALE*( (q+bu)·k[s] + bd(t, (T-1)-t+s) ); online softmax; @V.
// bd computed unshifted via WMMA at p-bases c0-15 / c0+1, shift-gathered in LDS.
// ============================================================================
#define KST 68   // k_lds row stride (halves); 136B keeps half4 stores aligned

__global__ __launch_bounds__(32)
void attn_kernel(const float* __restrict__ q, const float* __restrict__ k,
                 const float* __restrict__ v, const float* __restrict__ pbuf,
                 const float* __restrict__ bias_u, const float* __restrict__ bias_v,
                 float* __restrict__ out) {
  __shared__ __align__(16) float    q_lds[16*64];
  __shared__ float    bu_lds[64], bv_lds[64];
  __shared__ __align__(16) _Float16 k_lds[32*KST];   // [key][kd]
  __shared__ __align__(16) _Float16 vt_lds[64*34];   // transposed: [dk][key]
  __shared__ float    bd_lds[16*33];                 // unshifted bd strip (32 p cols)
  __shared__ float    sc_lds[16*34];                 // raw scores, 32 keys
  __shared__ __align__(16) _Float16 p16_lds[16*34];  // exp(scores) as f16
  __shared__ float    red_lds[32];
  __shared__ float    fac_lds[16];
  __shared__ float    l_lds[16];

  const int lane  = threadIdx.x;
  const int t0    = blockIdx.x * 16;
  const int head  = blockIdx.y;
  const int b     = blockIdx.z;
  const int khalf = lane >> 4;
  const int l16   = lane & 15;

  for (int i = lane; i < 256; i += 32) {            // 16x64 via float4
    int m = i >> 4, d4 = (i & 15) << 2;
    *(float4*)&q_lds[m*64 + d4] =
        *(const float4*)&q[((long)(b*T_ + t0 + m))*D_ + head*DK_ + d4];
  }
  bu_lds[lane]      = bias_u[head*DK_ + lane];
  bu_lds[lane + 32] = bias_u[head*DK_ + lane + 32];
  bv_lds[lane]      = bias_v[head*DK_ + lane];
  bv_lds[lane + 32] = bias_v[head*DK_ + lane + 32];
  __syncthreads();

  // q+bias_u / q+bias_v A-fragments, two K=32 steps covering DK=64
  v16h a_u[2], a_v[2];
  #pragma unroll
  for (int ks = 0; ks < 2; ++ks) {
    #pragma unroll
    for (int vv = 0; vv < 8; ++vv) {
      int kb = 2*vv + (vv >= 4 ? 8 : 0) + khalf*8;
      int kg = ks*32 + kb;
      float x0 = q_lds[l16*64 + kg], x1 = q_lds[l16*64 + kg + 1];
      a_u[ks][2*vv]   = (_Float16)(x0 + bu_lds[kg]);
      a_u[ks][2*vv+1] = (_Float16)(x1 + bu_lds[kg+1]);
      a_v[ks][2*vv]   = (_Float16)(x0 + bv_lds[kg]);
      a_v[ks][2*vv+1] = (_Float16)(x1 + bv_lds[kg+1]);
    }
  }

  float m_run = -1e30f, l_run = 0.0f;
  v8f ctx0 = {}, ctx1 = {}, ctx2 = {}, ctx3 = {};

  for (int j = 0; j < T_/32; ++j) {
    const int s0 = j*32;
    for (int i = lane; i < 512; i += 32) {          // 32x64 via float4
      int key = i >> 4, d4 = (i & 15) << 2;
      long base = ((long)(b*T_ + s0 + key))*D_ + head*DK_ + d4;
      float4 kv = *(const float4*)&k[base];
      float4 vv = *(const float4*)&v[base];
      h4 kh; kh[0]=(_Float16)kv.x; kh[1]=(_Float16)kv.y; kh[2]=(_Float16)kv.z; kh[3]=(_Float16)kv.w;
      *(h4*)&k_lds[key*KST + d4] = kh;
      vt_lds[(d4+0)*34 + key] = (_Float16)vv.x;
      vt_lds[(d4+1)*34 + key] = (_Float16)vv.y;
      vt_lds[(d4+2)*34 + key] = (_Float16)vv.z;
      vt_lds[(d4+3)*34 + key] = (_Float16)vv.w;
    }
    if (j + 1 < T_/32) {   // L2 hint for next chunk (global_prefetch_b8)
      long nb = ((long)(b*T_ + s0 + 32 + lane))*D_ + head*DK_;
      __builtin_prefetch(&k[nb], 0, 1);
      __builtin_prefetch(&v[nb], 0, 1);
    }
    __syncthreads();

    #pragma unroll
    for (int hh = 0; hh < 2; ++hh) {
      const int s0h = s0 + 16*hh;
      // ac = (q+bu) @ k^T  (16x16, two K steps)
      v8f acc = {};
      #pragma unroll
      for (int ks = 0; ks < 2; ++ks) {
        v16h bf;
        #pragma unroll
        for (int vv = 0; vv < 8; ++vv) {
          int kb = ks*32 + 2*vv + (vv >= 4 ? 8 : 0) + khalf*8;
          frag_set_pair(bf, vv, &k_lds[(16*hh + l16)*KST + kb]);
        }
        acc = __builtin_amdgcn_wmma_f32_16x16x32_f16(false, a_u[ks], false, bf, (short)0, acc, false, false);
      }
      // bd: two unshifted 16x16 tiles covering p in [c0-15, c0+16]
      const int pb0 = (T_ - 1) - t0 + s0h - 15;     // always in [0, 2047-31]
      #pragma unroll
      for (int pt = 0; pt < 2; ++pt) {
        v8f bdacc = {};
        const float* prow = pbuf + (long)(pb0 + 16*pt + l16)*D_ + head*DK_;
        #pragma unroll
        for (int ks = 0; ks < 2; ++ks) {
          v16h bf;
          #pragma unroll
          for (int vv = 0; vv < 8; ++vv) {
            int kb = ks*32 + 2*vv + (vv >= 4 ? 8 : 0) + khalf*8;
            bf[2*vv]   = (_Float16)prow[kb];
            bf[2*vv+1] = (_Float16)prow[kb+1];
          }
          bdacc = __builtin_amdgcn_wmma_f32_16x16x32_f16(false, a_v[ks], false, bf, (short)0, bdacc, false, false);
        }
        #pragma unroll
        for (int r = 0; r < 8; ++r)
          bd_lds[(r + 8*khalf)*33 + 16*pt + l16] = bdacc[r];
      }
      __syncthreads();
      // shift-gather: bd_shifted(m,n) = bd_lds[m][15 - m + n]
      #pragma unroll
      for (int r = 0; r < 8; ++r) {
        int m = r + 8*khalf;
        float sc = SCALE_*(acc[r] + bd_lds[m*33 + 15 - m + l16]);
        sc_lds[m*34 + 16*hh + l16] = sc;
      }
      __syncthreads();
    }

    // ---- online softmax across the 32 fresh keys (row = l16, half = khalf) --
    float lm = -1e30f;
    for (int cc = 0; cc < 16; ++cc) lm = fmaxf(lm, sc_lds[l16*34 + khalf*16 + cc]);
    red_lds[khalf*16 + l16] = lm;
    __syncthreads();
    float m_new = fmaxf(m_run, fmaxf(red_lds[l16], red_lds[16 + l16]));
    float fac = __expf(m_run - m_new);
    float ls = 0.f;
    for (int cc = 0; cc < 16; ++cc) {
      float e = __expf(sc_lds[l16*34 + khalf*16 + cc] - m_new);
      p16_lds[l16*34 + khalf*16 + cc] = (_Float16)e;
      ls += e;
    }
    __syncthreads();
    red_lds[khalf*16 + l16] = ls;
    if (khalf == 0) fac_lds[l16] = fac;
    __syncthreads();
    l_run = l_run*fac + red_lds[l16] + red_lds[16 + l16];
    m_run = m_new;

    // rescale running context, then ctx += probs(16x32) @ V(32x64)
    #pragma unroll
    for (int r = 0; r < 8; ++r) {
      float f = fac_lds[r + 8*khalf];
      ctx0[r]*=f; ctx1[r]*=f; ctx2[r]*=f; ctx3[r]*=f;
    }
    v16h ap;
    #pragma unroll
    for (int vv = 0; vv < 8; ++vv) {
      int kb = 2*vv + (vv >= 4 ? 8 : 0) + khalf*8;
      frag_set_pair(ap, vv, &p16_lds[l16*34 + kb]);
    }
    #pragma unroll
    for (int g = 0; g < 4; ++g) {
      v16h bf;
      #pragma unroll
      for (int vv = 0; vv < 8; ++vv) {
        int kb = 2*vv + (vv >= 4 ? 8 : 0) + khalf*8;
        frag_set_pair(bf, vv, &vt_lds[(16*g + l16)*34 + kb]);
      }
      v8f& cg = g==0?ctx0:(g==1?ctx1:(g==2?ctx2:ctx3));
      cg = __builtin_amdgcn_wmma_f32_16x16x32_f16(false, ap, false, bf, (short)0, cg, false, false);
    }
    __syncthreads();
  }

  if (khalf == 0) l_lds[l16] = l_run;
  __syncthreads();
  #pragma unroll
  for (int r = 0; r < 8; ++r) {
    int m = r + 8*khalf;
    float inv = 1.0f / l_lds[m];
    long base = ((long)(b*T_ + t0 + m))*D_ + head*DK_;
    out[base +  0 + l16] = ctx0[r]*inv;
    out[base + 16 + l16] = ctx1[r]*inv;
    out[base + 32 + l16] = ctx2[r]*inv;
    out[base + 48 + l16] = ctx3[r]*inv;
  }
}

// ============================================================================
// LayerNorm over D=512 (one block per row; safe in-place)
// ============================================================================
__global__ __launch_bounds__(256)
void ln_kernel(const float* __restrict__ in, const float* __restrict__ s,
               const float* __restrict__ bta, float* __restrict__ out) {
  __shared__ float red[256];
  const int row = blockIdx.x, tid = threadIdx.x;
  const float* xr = in + (long)row*D_;
  float v0 = xr[tid], v1 = xr[tid + 256];
  red[tid] = v0 + v1; __syncthreads();
  for (int st = 128; st > 0; st >>= 1) { if (tid < st) red[tid] += red[tid+st]; __syncthreads(); }
  float mean = red[0] * (1.0f/(float)D_);
  __syncthreads();
  float d0 = v0 - mean, d1 = v1 - mean;
  red[tid] = d0*d0 + d1*d1; __syncthreads();
  for (int st = 128; st > 0; st >>= 1) { if (tid < st) red[tid] += red[tid+st]; __syncthreads(); }
  float rstd = rsqrtf(red[0]*(1.0f/(float)D_) + EPS_);
  out[(long)row*D_ + tid]       = d0*rstd*s[tid]       + bta[tid];
  out[(long)row*D_ + tid + 256] = d1*rstd*s[tid + 256] + bta[tid + 256];
}

// ============================================================================
// GLU + depthwise conv (K=31, same-pad). One block per (b, channel d).
// ============================================================================
__global__ __launch_bounds__(256)
void dwconv_kernel(const float* __restrict__ y2, const float* __restrict__ w,
                   const float* __restrict__ bias, float* __restrict__ out) {
  __shared__ float g_lds[T_ + KCONV - 1];
  __shared__ float w_lds[KCONV];
  const int b = blockIdx.x / D_, d = blockIdx.x % D_, tid = threadIdx.x;
  if (tid < KCONV) w_lds[tid] = w[d*KCONV + tid];
  const int HALO = (KCONV-1)/2;
  for (int t = tid; t < T_ + 2*HALO; t += 256) {
    int ts = t - HALO;
    float val = 0.f;
    if (ts >= 0 && ts < T_) {
      long base = ((long)(b*T_ + ts))*(2*D_);
      float a = y2[base + d], gg = y2[base + D_ + d];
      val = a * sigmoidf_(gg);
    }
    g_lds[t] = val;
  }
  __syncthreads();
  float bs = bias[d];
  for (int t = tid; t < T_; t += 256) {
    float acc = bs;
    #pragma unroll
    for (int kk = 0; kk < KCONV; ++kk) acc += g_lds[t + kk] * w_lds[kk];
    out[((long)(b*T_ + t))*D_ + d] = acc;
  }
}

// BatchNorm batch statistics: one block per channel d
__global__ __launch_bounds__(256)
void bnstats_kernel(const float* __restrict__ y, float* __restrict__ mu,
                    float* __restrict__ rstd) {
  __shared__ float r1[256], r2[256];
  const int d = blockIdx.x, tid = threadIdx.x;
  float s = 0.f, s2 = 0.f;
  for (int e = tid; e < NROW; e += 256) { float t = y[(long)e*D_ + d]; s += t; s2 += t*t; }
  r1[tid] = s; r2[tid] = s2; __syncthreads();
  for (int st = 128; st > 0; st >>= 1) {
    if (tid < st) { r1[tid] += r1[tid+st]; r2[tid] += r2[tid+st]; }
    __syncthreads();
  }
  if (tid == 0) {
    float m = r1[0]*(1.0f/(float)NROW);
    float var = r2[0]*(1.0f/(float)NROW) - m*m;
    mu[d] = m; rstd[d] = rsqrtf(var + EPS_);
  }
}

__global__ void bnapply_kernel(const float* __restrict__ y, const float* __restrict__ mu,
                               const float* __restrict__ rstd, const float* __restrict__ g,
                               const float* __restrict__ bb, float* __restrict__ out, int n) {
  int i = blockIdx.x*blockDim.x + threadIdx.x;
  if (i >= n) return;
  int d = i % D_;
  float v = (y[i] - mu[d])*rstd[d]*g[d] + bb[d];
  out[i] = siluf_(v);
}

__global__ void fill_zero_kernel(float* p, int n) {
  int i = blockIdx.x*blockDim.x + threadIdx.x;
  if (i < n) p[i] = 0.f;
}

// ============================================================================
static void launch_gemm(const float* A, const float* W, const float* bias,
                        const float* res, float* out, int M, int N, int Kd,
                        int Arows, int act, float alpha, hipStream_t s) {
  dim3 grid(N/BN, M/BM);
  wmma_gemm_kernel<<<grid, 256, 0, s>>>(A, W, bias, res, out, M, N, Kd, Arows, act, alpha);
}

extern "C" void kernel_launch(void* const* d_in, const int* in_sizes, int n_in,
                              void* d_out, int out_size, void* d_ws, size_t ws_size,
                              hipStream_t stream) {
  (void)in_sizes; (void)n_in; (void)out_size; (void)ws_size;
  const float* in_x   = (const float*)d_in[0];
  const float* pos    = (const float*)d_in[1];   // (1,P,D)
  const float* ln_s   = (const float*)d_in[2];
  const float* ln_b   = (const float*)d_in[3];
  const float* ff1_w1 = (const float*)d_in[4];
  const float* ff1_b1 = (const float*)d_in[5];
  const float* ff1_w2 = (const float*)d_in[6];
  const float* ff1_b2 = (const float*)d_in[7];
  const float* ff2_w1 = (const float*)d_in[8];
  const float* ff2_b1 = (const float*)d_in[9];
  const float* ff2_w2 = (const float*)d_in[10];
  const float* ff2_b2 = (const float*)d_in[11];
  const float* wq     = (const float*)d_in[12];
  const float* bq     = (const float*)d_in[13];
  const float* wk     = (const float*)d_in[14];
  const float* bk     = (const float*)d_in[15];
  const float* wv     = (const float*)d_in[16];
  const float* bv     = (const float*)d_in[17];
  const float* wp     = (const float*)d_in[18];
  const float* wo     = (const float*)d_in[19];
  const float* bo     = (const float*)d_in[20];
  const float* bias_u = (const float*)d_in[21];
  const float* bias_v = (const float*)d_in[22];
  const float* pw1_w  = (const float*)d_in[23];
  const float* pw1_b  = (const float*)d_in[24];
  const float* dw_w   = (const float*)d_in[25];
  const float* dw_b   = (const float*)d_in[26];
  const float* bn_g   = (const float*)d_in[27];
  const float* bn_b   = (const float*)d_in[28];
  const float* pw2_w  = (const float*)d_in[29];
  const float* pw2_b  = (const float*)d_in[30];

  // workspace carve-up (floats)
  float* ws = (float*)d_ws;
  size_t off = 0;
  float* x    = ws + off; off += (size_t)NROW*D_;
  float* h    = ws + off; off += (size_t)NROW*D_;
  float* qb   = ws + off; off += (size_t)NROW*D_;
  float* kbuf = ws + off; off += (size_t)NROW*D_;
  float* vbuf = ws + off; off += (size_t)NROW*D_;
  float* pb   = ws + off; off += (size_t)PPAD*D_;
  float* tmp  = ws + off; off += (size_t)NROW*DFF_;
  float* y2   = ws + off; off += (size_t)NROW*2*D_;
  float* yc   = ws + off; off += (size_t)NROW*D_;
  float* mu   = ws + off; off += 512;
  float* rstd = ws + off; off += 512;

  hipMemcpyAsync(x, in_x, sizeof(float)*(size_t)NROW*D_, hipMemcpyDeviceToDevice, stream);

  for (int l = 0; l < L_; ++l) {
    const float* LS = ln_s + (size_t)l*5*D_;
    const float* LB = ln_b + (size_t)l*5*D_;

    // --- macaron FFN 1 (half residual) ---
    ln_kernel<<<NROW, 256, 0, stream>>>(x, LS + 0*D_, LB + 0*D_, h);
    launch_gemm(h,   ff1_w1 + (size_t)l*D_*DFF_, ff1_b1 + (size_t)l*DFF_, nullptr, tmp, NROW, DFF_, D_,   NROW, 1, 1.0f, stream);
    launch_gemm(tmp, ff1_w2 + (size_t)l*DFF_*D_, ff1_b2 + (size_t)l*D_,   x,       x,   NROW, D_,   DFF_, NROW, 0, 0.5f, stream);

    // --- rel-pos MHSA ---
    ln_kernel<<<NROW, 256, 0, stream>>>(x, LS + 1*D_, LB + 1*D_, h);
    launch_gemm(h,   wq + (size_t)l*D_*D_, bq + (size_t)l*D_, nullptr, qb,   NROW, D_, D_, NROW, 0, 1.0f, stream);
    launch_gemm(h,   wk + (size_t)l*D_*D_, bk + (size_t)l*D_, nullptr, kbuf, NROW, D_, D_, NROW, 0, 1.0f, stream);
    launch_gemm(h,   wv + (size_t)l*D_*D_, bv + (size_t)l*D_, nullptr, vbuf, NROW, D_, D_, NROW, 0, 1.0f, stream);
    // position projection: run padded to 2048 rows, A reads clamped to P_-1,
    // then re-zero pad row 2047 (bd edge tiles require it to be zero)
    launch_gemm(pos, wp + (size_t)l*D_*D_, nullptr, nullptr, pb, PPAD, D_, D_, P_, 0, 1.0f, stream);
    fill_zero_kernel<<<2, 256, 0, stream>>>(pb + (size_t)P_*D_, D_);
    attn_kernel<<<dim3(T_/16, H_, B_), 32, 0, stream>>>(qb, kbuf, vbuf, pb,
        bias_u + (size_t)l*H_*DK_, bias_v + (size_t)l*H_*DK_, h);
    launch_gemm(h, wo + (size_t)l*D_*D_, bo + (size_t)l*D_, x, x, NROW, D_, D_, NROW, 0, 1.0f, stream);

    // --- conv module ---
    ln_kernel<<<NROW, 256, 0, stream>>>(x, LS + 2*D_, LB + 2*D_, h);
    launch_gemm(h, pw1_w + (size_t)l*D_*2*D_, pw1_b + (size_t)l*2*D_, nullptr, y2, NROW, 2*D_, D_, NROW, 0, 1.0f, stream);
    dwconv_kernel<<<B_*D_, 256, 0, stream>>>(y2, dw_w + (size_t)l*D_*KCONV, dw_b + (size_t)l*D_, yc);
    bnstats_kernel<<<D_, 256, 0, stream>>>(yc, mu, rstd);
    bnapply_kernel<<<((size_t)NROW*D_ + 255)/256, 256, 0, stream>>>(yc, mu, rstd,
        bn_g + (size_t)l*D_, bn_b + (size_t)l*D_, h, NROW*D_);
    launch_gemm(h, pw2_w + (size_t)l*D_*D_, pw2_b + (size_t)l*D_, x, x, NROW, D_, D_, NROW, 0, 1.0f, stream);

    // --- macaron FFN 2 (half residual) ---
    ln_kernel<<<NROW, 256, 0, stream>>>(x, LS + 3*D_, LB + 3*D_, h);
    launch_gemm(h,   ff2_w1 + (size_t)l*D_*DFF_, ff2_b1 + (size_t)l*DFF_, nullptr, tmp, NROW, DFF_, D_,   NROW, 1, 1.0f, stream);
    launch_gemm(tmp, ff2_w2 + (size_t)l*DFF_*D_, ff2_b2 + (size_t)l*D_,   x,       x,   NROW, D_,   DFF_, NROW, 0, 0.5f, stream);

    // --- final norm (last layer writes straight to d_out) ---
    float* lnout = (l == L_ - 1) ? (float*)d_out : x;
    ln_kernel<<<NROW, 256, 0, stream>>>(x, LS + 4*D_, LB + 4*D_, lnout);
  }
}
